// AspectLinkModel_47261820125865
// MI455X (gfx1250) — compile-verified, matrix operand order checked by reading, not run on hardware
//
#include <hip/hip_runtime.h>
#include <hip/hip_bf16.h>

// ---------------------------------------------------------------------------
// Types for CDNA5 WMMA (wave32): v_wmma_f32_16x16x32_bf16
// ---------------------------------------------------------------------------
typedef __attribute__((ext_vector_type(16))) __bf16 v16bf;
typedef __attribute__((ext_vector_type(2)))  __bf16 v2bf;
typedef __attribute__((ext_vector_type(8)))  float  v8f;
typedef __attribute__((ext_vector_type(4)))  unsigned short us4;

__device__ __forceinline__ unsigned short f2bf(float x) {
    unsigned int u = __builtin_bit_cast(unsigned int, x);
    unsigned int r = u + 0x7FFFu + ((u >> 16) & 1u);   // round-to-nearest-even
    return (unsigned short)(r >> 16);
}
__device__ __forceinline__ us4 f2bf4(float4 v) {
    us4 r;
    r[0] = f2bf(v.x); r[1] = f2bf(v.y); r[2] = f2bf(v.z); r[3] = f2bf(v.w);
    return r;
}

__device__ __forceinline__ float gelu_tanh(float x) {
    const float c = 0.7978845608028654f;   // sqrt(2/pi)
    float x3 = x * x * x;
    return 0.5f * x * (1.0f + tanhf(c * (x + 0.044715f * x3)));
}

// ---------------------------------------------------------------------------
// Block reductions (blockDim.x == 256)
// ---------------------------------------------------------------------------
__device__ __forceinline__ float blockSum(float v, float* red) {
    int tid = threadIdx.x;
    red[tid] = v; __syncthreads();
    for (int s = 128; s > 0; s >>= 1) {
        if (tid < s) red[tid] += red[tid + s];
        __syncthreads();
    }
    float r = red[0]; __syncthreads();
    return r;
}
__device__ __forceinline__ float blockMax(float v, float* red) {
    int tid = threadIdx.x;
    red[tid] = v; __syncthreads();
    for (int s = 128; s > 0; s >>= 1) {
        if (tid < s) red[tid] = fmaxf(red[tid], red[tid + s]);
        __syncthreads();
    }
    float r = red[0]; __syncthreads();
    return r;
}

// ---------------------------------------------------------------------------
// WMMA GEMM:  C[M,N] = act( scale * A[M,K] @ B + bias )
//   TRANSB=false : B row-major [K,N], ldb   (weights)
//   TRANSB=true  : B row-major [N,K], ldb   (A @ B^T)
// Batched via gridDim.z:  z -> (b2 = z/batH, h2 = z%batH) with element strides.
// Block: 256 threads = 8 wave32, C tile 128x64, each wave 32x32 (2x2 WMMA).
// Software-pipelined: next K-tile global loads overlap fragment DS + WMMA.
// ---------------------------------------------------------------------------
#define BM 128
#define BN 64
#define BK 32

template<bool TRANSB, bool GELU>
__global__ __launch_bounds__(256)
void gemm_wmma(const float* __restrict__ A, const float* __restrict__ B,
               float* __restrict__ C, const float* __restrict__ bias,
               int M, int N, int K, int lda, int ldb, int ldc,
               int batH, long sAb, long sAh, long sBb, long sBh,
               long sCb, long sCh, float scale)
{
    int z  = blockIdx.z;
    int b2 = z / batH, h2 = z % batH;
    A += b2 * sAb + h2 * sAh;
    B += b2 * sBb + h2 * sBh;
    C += b2 * sCb + h2 * sCh;

    __shared__ unsigned short As[BM][BK + 4];   // bf16 bits, [m][k]
    __shared__ unsigned short Bs[BN][BK + 4];   // bf16 bits, [n][k]

    const int tid  = threadIdx.x;
    const int wave = tid >> 5;
    const int lane = tid & 31;
    const int half = lane >> 4;
    const int l16  = lane & 15;

    const int wm = (wave >> 1) * 32;
    const int wn = (wave & 1) * 32;
    const int m0 = blockIdx.x * BM;
    const int n0 = blockIdx.y * BN;

    const bool fullA = (m0 + BM <= M);   // uniform across block
    const bool fullB = (n0 + BN <= N);   // uniform across block

    // Per-thread staging coordinates (fixed across K-steps):
    //   A: 4 float4 vectors; vector idx = e*256+tid in [0,1024)
    //      row = idx>>3 (8 vec/row), col = (idx&7)*4
    //   B: 2 float4 vectors; TRANSB: idx>>3 = n-row (8 vec/row of BK)
    //      non-TRANSB: idx>>4 = k-row (16 vec/row of BN), n4=(idx&15)*4
    float4 ra[4], rb[2];

    auto loadA = [&](int k0, float4* r) {
        const bool kOK = (k0 + BK <= K);
        if (fullA && kOK) {
            #pragma unroll
            for (int e = 0; e < 4; ++e) {
                int idx = e * 256 + tid;
                int row = idx >> 3, col = (idx & 7) * 4;
                r[e] = *(const float4*)(A + (long)(m0 + row) * lda + k0 + col);
            }
        } else {
            #pragma unroll
            for (int e = 0; e < 4; ++e) {
                int idx = e * 256 + tid;
                int row = idx >> 3, col = (idx & 7) * 4;
                float tmp[4];
                #pragma unroll
                for (int j = 0; j < 4; ++j) {
                    int gm = m0 + row, gk = k0 + col + j;
                    tmp[j] = (gm < M && gk < K) ? A[(long)gm * lda + gk] : 0.0f;
                }
                r[e] = make_float4(tmp[0], tmp[1], tmp[2], tmp[3]);
            }
        }
    };
    auto loadB = [&](int k0, float4* r) {
        const bool kOK = (k0 + BK <= K);
        if (fullB && kOK) {
            #pragma unroll
            for (int e = 0; e < 2; ++e) {
                int idx = e * 256 + tid;
                if (TRANSB) {
                    int n = idx >> 3, col = (idx & 7) * 4;
                    r[e] = *(const float4*)(B + (long)(n0 + n) * ldb + k0 + col);
                } else {
                    int k = idx >> 4, n4 = (idx & 15) * 4;
                    r[e] = *(const float4*)(B + (long)(k0 + k) * ldb + n0 + n4);
                }
            }
        } else {
            #pragma unroll
            for (int e = 0; e < 2; ++e) {
                int idx = e * 256 + tid;
                float tmp[4];
                #pragma unroll
                for (int j = 0; j < 4; ++j) {
                    int gk, gn;
                    if (TRANSB) { gn = n0 + (idx >> 3); gk = k0 + (idx & 7) * 4 + j; }
                    else        { gk = k0 + (idx >> 4); gn = n0 + (idx & 15) * 4 + j; }
                    float v = 0.0f;
                    if (gn < N && gk < K)
                        v = TRANSB ? B[(long)gn * ldb + gk] : B[(long)gk * ldb + gn];
                    tmp[j] = v;
                }
                r[e] = make_float4(tmp[0], tmp[1], tmp[2], tmp[3]);
            }
        }
    };
    auto storeLDS = [&](const float4* a, const float4* b) {
        #pragma unroll
        for (int e = 0; e < 4; ++e) {
            int idx = e * 256 + tid;
            int row = idx >> 3, col = (idx & 7) * 4;
            *(us4*)&As[row][col] = f2bf4(a[e]);
        }
        #pragma unroll
        for (int e = 0; e < 2; ++e) {
            int idx = e * 256 + tid;
            if (TRANSB) {
                int n = idx >> 3, col = (idx & 7) * 4;
                *(us4*)&Bs[n][col] = f2bf4(b[e]);
            } else {
                int k = idx >> 4, n4 = (idx & 15) * 4;
                us4 p = f2bf4(b[e]);
                Bs[n4 + 0][k] = p[0]; Bs[n4 + 1][k] = p[1];
                Bs[n4 + 2][k] = p[2]; Bs[n4 + 3][k] = p[3];
            }
        }
    };

    v8f acc[2][2] = {};

    loadA(0, ra);
    loadB(0, rb);

    for (int k0 = 0; k0 < K; k0 += BK) {
        storeLDS(ra, rb);
        __syncthreads();

        // Prefetch next K-tile while this tile computes (overlap HBM/L2 latency)
        float4 na[4], nb[2];
        const bool more = (k0 + BK < K);
        if (more) { loadA(k0 + BK, na); loadB(k0 + BK, nb); }

        // ---- build fragments per documented 16-bit A/B VGPR layout ----
        // element pair p (VGPR p): K base = ((p>>2)<<4) + ((p&3)<<1) + 8*half
        v16bf aF[2], bF[2];
        #pragma unroll
        for (int p = 0; p < 8; ++p) {
            int kb = ((p >> 2) << 4) + ((p & 3) << 1) + (half << 3);
            #pragma unroll
            for (int s = 0; s < 2; ++s) {
                unsigned int ua = *(const unsigned int*)&As[wm + s * 16 + l16][kb];
                v2bf pa = __builtin_bit_cast(v2bf, ua);
                aF[s][2 * p] = pa[0]; aF[s][2 * p + 1] = pa[1];
                unsigned int ub = *(const unsigned int*)&Bs[wn + s * 16 + l16][kb];
                v2bf pb = __builtin_bit_cast(v2bf, ub);
                bF[s][2 * p] = pb[0]; bF[s][2 * p + 1] = pb[1];
            }
        }
        #pragma unroll
        for (int mi = 0; mi < 2; ++mi)
            #pragma unroll
            for (int ni = 0; ni < 2; ++ni)
                acc[mi][ni] = __builtin_amdgcn_wmma_f32_16x16x32_bf16(
                    false, aF[mi], false, bF[ni], (short)0, acc[mi][ni],
                    false, false);
        __syncthreads();

        if (more) {
            #pragma unroll
            for (int e = 0; e < 4; ++e) ra[e] = na[e];
            #pragma unroll
            for (int e = 0; e < 2; ++e) rb[e] = nb[e];
        }
    }

    // ---- epilogue: scale, bias, activation, store ----
    if (fullA && fullB) {
        #pragma unroll
        for (int mi = 0; mi < 2; ++mi) {
            #pragma unroll
            for (int ni = 0; ni < 2; ++ni) {
                int ng = n0 + wn + ni * 16 + l16;
                float bv = (bias != nullptr) ? bias[ng] : 0.0f;
                #pragma unroll
                for (int r = 0; r < 8; ++r) {
                    int mg = m0 + wm + mi * 16 + r + half * 8;
                    float v = acc[mi][ni][r] * scale + bv;
                    if (GELU) v = gelu_tanh(v);
                    C[(long)mg * ldc + ng] = v;
                }
            }
        }
    } else {
        #pragma unroll
        for (int mi = 0; mi < 2; ++mi) {
            #pragma unroll
            for (int ni = 0; ni < 2; ++ni) {
                int ng = n0 + wn + ni * 16 + l16;
                float bv = (bias != nullptr && ng < N) ? bias[ng] : 0.0f;
                #pragma unroll
                for (int r = 0; r < 8; ++r) {
                    int mg = m0 + wm + mi * 16 + r + half * 8;
                    if (mg < M && ng < N) {
                        float v = acc[mi][ni][r] * scale + bv;
                        if (GELU) v = gelu_tanh(v);
                        C[(long)mg * ldc + ng] = v;
                    }
                }
            }
        }
    }
}

// ---------------------------------------------------------------------------
// h = LN(x + pos_emb[s]) * g + b     (one block per token, 768 features)
// ---------------------------------------------------------------------------
__global__ __launch_bounds__(256)
void add_pos_ln(const float* __restrict__ x, const float* __restrict__ pos,
                const float* __restrict__ g, const float* __restrict__ b,
                float* __restrict__ out, int S)
{
    __shared__ float red[256];
    int t = blockIdx.x, tid = threadIdx.x;
    int s = t % S;
    long base = (long)t * 768;
    float lv[3]; float sum = 0.0f;
    #pragma unroll
    for (int j = 0; j < 3; ++j) {
        int f = tid + j * 256;
        lv[j] = x[base + f] + pos[(long)s * 768 + f];
        sum += lv[j];
    }
    float mean = blockSum(sum, red) * (1.0f / 768.0f);
    float var = 0.0f;
    #pragma unroll
    for (int j = 0; j < 3; ++j) { float d = lv[j] - mean; var += d * d; }
    var = blockSum(var, red) * (1.0f / 768.0f);
    float inv = rsqrtf(var + 1e-12f);
    #pragma unroll
    for (int j = 0; j < 3; ++j) {
        int f = tid + j * 256;
        out[base + f] = (lv[j] - mean) * inv * g[f] + b[f];
    }
}

// ---------------------------------------------------------------------------
// h = LN(h + y) * g + b ; optionally copy result into concat buffer
// ---------------------------------------------------------------------------
__global__ __launch_bounds__(256)
void residual_ln(float* __restrict__ h, const float* __restrict__ y,
                 const float* __restrict__ g, const float* __restrict__ b,
                 float* __restrict__ cat, int catOff, int catLd)
{
    __shared__ float red[256];
    int t = blockIdx.x, tid = threadIdx.x;
    long base = (long)t * 768;
    float lv[3]; float sum = 0.0f;
    #pragma unroll
    for (int j = 0; j < 3; ++j) {
        int f = tid + j * 256;
        lv[j] = h[base + f] + y[base + f];
        sum += lv[j];
    }
    float mean = blockSum(sum, red) * (1.0f / 768.0f);
    float var = 0.0f;
    #pragma unroll
    for (int j = 0; j < 3; ++j) { float d = lv[j] - mean; var += d * d; }
    var = blockSum(var, red) * (1.0f / 768.0f);
    float inv = rsqrtf(var + 1e-12f);
    #pragma unroll
    for (int j = 0; j < 3; ++j) {
        int f = tid + j * 256;
        float o = (lv[j] - mean) * inv * g[f] + b[f];
        h[base + f] = o;
        if (cat != nullptr) cat[(long)t * catLd + catOff + f] = o;
    }
}

// ---------------------------------------------------------------------------
// In-place row softmax, row length n <= 256, one block per row
// ---------------------------------------------------------------------------
__global__ __launch_bounds__(256)
void softmax_rows(float* __restrict__ x, int n)
{
    __shared__ float red[256];
    long row = blockIdx.x;
    float* p = x + row * (long)n;
    int tid = threadIdx.x;
    float v = (tid < n) ? p[tid] : -INFINITY;
    float m = blockMax(v, red);
    float e = (tid < n) ? __expf(v - m) : 0.0f;
    float s = blockSum(e, red);
    if (tid < n) p[tid] = e / s;
}

// ---------------------------------------------------------------------------
// out[b,:] = sum over rows s of  x[b,s,:] / max(||x[b,s,:]||, 1e-8)
// ---------------------------------------------------------------------------
__global__ __launch_bounds__(256)
void norm_sum(const float* __restrict__ x, float* __restrict__ out, int S)
{
    __shared__ float red[256];
    int b = blockIdx.x, tid = threadIdx.x;
    const int F = 3072;
    float acc[12];
    #pragma unroll
    for (int j = 0; j < 12; ++j) acc[j] = 0.0f;
    for (int s = 0; s < S; ++s) {
        const float* row = x + ((long)b * S + s) * F;
        float vals[12]; float ss = 0.0f;
        #pragma unroll
        for (int j = 0; j < 12; ++j) {
            vals[j] = row[tid + j * 256];
            ss += vals[j] * vals[j];
        }
        float tot = blockSum(ss, red);
        float inv = 1.0f / fmaxf(sqrtf(tot), 1e-8f);
        #pragma unroll
        for (int j = 0; j < 12; ++j) acc[j] += vals[j] * inv;
    }
    #pragma unroll
    for (int j = 0; j < 12; ++j) out[(long)b * F + tid + j * 256] = acc[j];
}

// out[b] = dot(sa[b,:], sc[b,:]) over 3072 features
__global__ __launch_bounds__(256)
void final_dot(const float* __restrict__ sa, const float* __restrict__ sc,
               float* __restrict__ out)
{
    __shared__ float red[256];
    int b = blockIdx.x, tid = threadIdx.x;
    float s = 0.0f;
    #pragma unroll
    for (int j = 0; j < 12; ++j) {
        int f = tid + j * 256;
        s += sa[(long)b * 3072 + f] * sc[(long)b * 3072 + f];
    }
    float tot = blockSum(s, red);
    if (tid == 0) out[b] = tot;
}

// ---------------------------------------------------------------------------
// Host-side encoder driver
// ---------------------------------------------------------------------------
struct EncW {
    const float *fc1_w, *fc1_b, *pos, *emb_g, *emb_b;
    const float *Wq, *bq, *Wk, *bk, *Wv, *bv, *Wo, *bo;
    const float *ln1g, *ln1b, *W1, *bf1, *W2, *bf2, *ln2g, *ln2b;
};

static void run_encoder(const float* x, int Bn, int S, const EncW& E,
                        float* h, float* q, float* k, float* v,
                        float* attno, float* proj, float* big, float* cat,
                        hipStream_t stream)
{
    const int D = 768, FF = 3072, H = 12, dh = 64;
    const int T = Bn * S;
    dim3 blk(256);
    auto gM = [](int M) { return (unsigned)((M + BM - 1) / BM); };
    auto gN = [](int N) { return (unsigned)((N + BN - 1) / BN); };
    const long SS = (long)S * S;

    // fc1: x[T,300] @ fc1_w[300,768] + b
    gemm_wmma<false, false><<<dim3(gM(T), gN(D), 1), blk, 0, stream>>>(
        x, E.fc1_w, proj, E.fc1_b, T, D, 300, 300, D, D,
        1, 0, 0, 0, 0, 0, 0, 1.0f);
    add_pos_ln<<<dim3(T), blk, 0, stream>>>(proj, E.pos, E.emb_g, E.emb_b, h, S);

    for (int l = 0; l < 4; ++l) {
        const long WD = (long)l * D * D;
        // Q, K, V projections
        gemm_wmma<false, false><<<dim3(gM(T), gN(D), 1), blk, 0, stream>>>(
            h, E.Wq + WD, q, E.bq + l * D, T, D, D, D, D, D,
            1, 0, 0, 0, 0, 0, 0, 1.0f);
        gemm_wmma<false, false><<<dim3(gM(T), gN(D), 1), blk, 0, stream>>>(
            h, E.Wk + WD, k, E.bk + l * D, T, D, D, D, D, D,
            1, 0, 0, 0, 0, 0, 0, 1.0f);
        gemm_wmma<false, false><<<dim3(gM(T), gN(D), 1), blk, 0, stream>>>(
            h, E.Wv + WD, v, E.bv + l * D, T, D, D, D, D, D,
            1, 0, 0, 0, 0, 0, 0, 1.0f);
        // scores[b,hd] = q_slice @ k_slice^T / 8   (batched B*H)
        gemm_wmma<true, false><<<dim3(gM(S), gN(S), Bn * H), blk, 0, stream>>>(
            q, k, big, nullptr, S, S, dh, D, D, S,
            H, (long)S * D, dh, (long)S * D, dh, (long)H * SS, SS, 0.125f);
        softmax_rows<<<dim3((unsigned)(Bn * H * S)), blk, 0, stream>>>(big, S);
        // attno[b,:,hd*64..] = attn @ v_slice
        gemm_wmma<false, false><<<dim3(gM(S), gN(dh), Bn * H), blk, 0, stream>>>(
            big, v, attno, nullptr, S, dh, S, S, D, D,
            H, (long)H * SS, SS, (long)S * D, dh, (long)S * D, dh, 1.0f);
        // output projection + residual LN1
        gemm_wmma<false, false><<<dim3(gM(T), gN(D), 1), blk, 0, stream>>>(
            attno, E.Wo + WD, proj, E.bo + l * D, T, D, D, D, D, D,
            1, 0, 0, 0, 0, 0, 0, 1.0f);
        residual_ln<<<dim3(T), blk, 0, stream>>>(
            h, proj, E.ln1g + l * D, E.ln1b + l * D, nullptr, 0, 0);
        // FFN1 (+gelu) and FFN2 + residual LN2 (writes concat slice)
        gemm_wmma<false, true><<<dim3(gM(T), gN(FF), 1), blk, 0, stream>>>(
            h, E.W1 + (long)l * D * FF, big, E.bf1 + l * FF, T, FF, D, D, FF, FF,
            1, 0, 0, 0, 0, 0, 0, 1.0f);
        gemm_wmma<false, false><<<dim3(gM(T), gN(D), 1), blk, 0, stream>>>(
            big, E.W2 + (long)l * FF * D, proj, E.bf2 + l * D, T, D, FF, FF, D, D,
            1, 0, 0, 0, 0, 0, 0, 1.0f);
        residual_ln<<<dim3(T), blk, 0, stream>>>(
            h, proj, E.ln2g + l * D, E.ln2b + l * D, cat, (3 - l) * D, 4 * D);
    }
}

// ---------------------------------------------------------------------------
// kernel_launch
// ---------------------------------------------------------------------------
extern "C" void kernel_launch(void* const* d_in, const int* in_sizes, int n_in,
                              void* d_out, int out_size, void* d_ws, size_t ws_size,
                              hipStream_t stream)
{
    (void)in_sizes; (void)n_in; (void)out_size;
    const float* ctx = (const float*)d_in[0];   // [16,256,300]
    const float* asp = (const float*)d_in[1];   // [16,128,300]
    EncW E;
    E.fc1_w = (const float*)d_in[2];  E.fc1_b = (const float*)d_in[3];
    E.pos   = (const float*)d_in[4];
    E.emb_g = (const float*)d_in[5];  E.emb_b = (const float*)d_in[6];
    E.Wq = (const float*)d_in[7];   E.bq = (const float*)d_in[8];
    E.Wk = (const float*)d_in[9];   E.bk = (const float*)d_in[10];
    E.Wv = (const float*)d_in[11];  E.bv = (const float*)d_in[12];
    E.Wo = (const float*)d_in[13];  E.bo = (const float*)d_in[14];
    E.ln1g = (const float*)d_in[15]; E.ln1b = (const float*)d_in[16];
    E.W1 = (const float*)d_in[17];  E.bf1 = (const float*)d_in[18];
    E.W2 = (const float*)d_in[19];  E.bf2 = (const float*)d_in[20];
    E.ln2g = (const float*)d_in[21]; E.ln2b = (const float*)d_in[22];

    // ---- workspace carve (floats) ----
    const long Tc = 16L * 256, Ta = 16L * 128;
    const long nTok = Tc * 768;                 // 3,145,728
    const long nBig = 16L * 12 * 256 * 256;     // 12,582,912 (== Tc*3072)
    float* ws = (float*)d_ws;
    long need = 6 * nTok + nBig + Tc * 3072 + Ta * 3072 + 2 * 16L * 3072;
    if (ws_size < (size_t)need * sizeof(float)) return;

    float* h     = ws;               ws += nTok;
    float* q     = ws;               ws += nTok;
    float* k     = ws;               ws += nTok;
    float* v     = ws;               ws += nTok;
    float* attno = ws;               ws += nTok;
    float* proj  = ws;               ws += nTok;
    float* big   = ws;               ws += nBig;       // scores / ffn intermediate
    float* catC  = ws;               ws += Tc * 3072;
    float* catA  = ws;               ws += Ta * 3072;
    float* sumA  = ws;               ws += 16L * 3072;
    float* sumC  = ws;               ws += 16L * 3072;

    run_encoder(ctx, 16, 256, E, h, q, k, v, attno, proj, big, catC, stream);
    run_encoder(asp, 16, 128, E, h, q, k, v, attno, proj, big, catA, stream);

    // sum over all (a,c) cosine pairs factorizes:
    //   sum_{a,c} an.cn = (sum_a an) . (sum_c cn)
    norm_sum<<<dim3(16), dim3(256), 0, stream>>>(catA, sumA, 128);
    norm_sum<<<dim3(16), dim3(256), 0, stream>>>(catC, sumC, 256);
    final_dot<<<dim3(16), dim3(256), 0, stream>>>(sumA, sumC, (float*)d_out);
}
